// TensoSDF_3143916061277
// MI455X (gfx1250) — matrix operand order repaired and checked by reading, not compiled, not run on hardware
//
#include <hip/hip_runtime.h>
#include <hip/hip_fp16.h>
#include <stdint.h>

typedef __attribute__((ext_vector_type(16))) _Float16 v16h;
typedef __attribute__((ext_vector_type(8)))  _Float16 v8h;
typedef __attribute__((ext_vector_type(8)))  float    v8f;

#define NPTS      524288
#define GRIDW     512
#define NCOMP     36
#define PLANE_PIX (GRIDW*GRIDW)
#define SDF_DIM   256
#define IN_CH     129          // 3*36 + 21
#define OUT_CH    129          // 1 + 128
#define KPAD1     160          // IN_CH padded to 5*32
#define NT1       16           // SDF_DIM/16 column tiles, GEMM1
#define KC1       5            // KPAD1/32
#define NT2       9            // OUT_CH padded 144 -> 9 tiles
#define KC2       8            // SDF_DIM/32

#define W1_HALVES (KC1*NT1*32*16)     // 40960 halves = 80 KB
#define W2_HALVES (KC2*NT2*32*16)     // 36864 halves = 72 KB
#define STG_STRIDE 264                // 256 + 8 pad halves
#define WAVES_PER_BLOCK 16
#define PTS_PER_BLOCK  (WAVES_PER_BLOCK*16)   // 256
#define STG_HALVES (WAVES_PER_BLOCK*16*STG_STRIDE)
#define SMEM_HALVES (W1_HALVES + W2_HALVES + STG_HALVES)
#define SMEM_BYTES  (SMEM_HALVES*2)   // 290816 B < 320 KB WGP LDS

// ---------------------------------------------------------------------------
// Plane transpose: [3][36][512][512] f32 (channel-major) ->
//                  [3][512][512][36] f16 (channel-last) in workspace.
// ---------------------------------------------------------------------------
__global__ void tp_planes_kernel(const float* __restrict__ planes,
                                 _Float16* __restrict__ out) {
  __shared__ _Float16 tile[64 * NCOMP];
  const int i  = blockIdx.x / (PLANE_PIX / 64);
  const int pb = (blockIdx.x % (PLANE_PIX / 64)) * 64;
  for (int idx = threadIdx.x; idx < 64 * NCOMP; idx += blockDim.x) {
    const int c = idx >> 6, px = idx & 63;
    tile[px * NCOMP + c] =
        (_Float16)planes[(size_t)(i * NCOMP + c) * PLANE_PIX + pb + px];
  }
  __syncthreads();
  const size_t ob = (size_t)(i * PLANE_PIX + pb) * NCOMP;
  for (int idx = threadIdx.x; idx < 64 * NCOMP; idx += blockDim.x)
    out[ob + idx] = tile[idx];
}

// Lines: [3][36][512] f32 -> [3][512][36] f16 (tiny)
__global__ void tp_lines_kernel(const float* __restrict__ lines,
                                _Float16* __restrict__ out) {
  const int idx = blockIdx.x * blockDim.x + threadIdx.x;
  if (idx >= 3 * NCOMP * GRIDW) return;
  const int c = idx % NCOMP;
  const int l = (idx / NCOMP) % GRIDW;
  const int i = idx / (NCOMP * GRIDW);
  out[idx] = (_Float16)lines[(i * NCOMP + c) * GRIDW + l];
}

__device__ __forceinline__ v16h combine16(v8h lo, v8h hi) {
  v16h r;
#pragma unroll
  for (int j = 0; j < 8; ++j) { r[j] = lo[j]; r[8 + j] = hi[j]; }
  return r;
}

// ---------------------------------------------------------------------------
// Fused: feature gather -> h = softplus(100*(f@W1+b1))/100 -> out = h@W2+b2
// ---------------------------------------------------------------------------
template <bool TP>
__global__ __launch_bounds__(512, 1)
void tenso_fused_kernel(const float* __restrict__ xyz,
                        const float* __restrict__ planes,
                        const float* __restrict__ lines,
                        const _Float16* __restrict__ tpp,
                        const _Float16* __restrict__ tpl,
                        const float* __restrict__ W1,
                        const float* __restrict__ b1,
                        const float* __restrict__ W2,
                        const float* __restrict__ b2,
                        float* __restrict__ out) {
  extern __shared__ _Float16 smem[];
  _Float16* w1f = smem;                       // B fragments for GEMM1
  _Float16* w2f = smem + W1_HALVES;           // B fragments for GEMM2
  const int tid  = threadIdx.x;
  const int wave = tid >> 5;
  const int lane = tid & 31;
  _Float16* stg = smem + W1_HALVES + W2_HALVES + wave * (16 * STG_STRIDE);

  // ---- zero weight-fragment LDS (covers K/N padding) ----
  {
    uint32_t* z = (uint32_t*)smem;
    for (int i = tid; i < (W1_HALVES + W2_HALVES) / 2; i += 512) z[i] = 0u;
  }
  __syncthreads();

  // ---- stage W1 [129,256] into B-fragment layout (table-order K, matching A) ----
  for (int idx = tid; idx < IN_CH * SDF_DIM; idx += 512) {
    const int k = idx >> 8, n = idx & 255;
    const int c = k >> 5, j = (k & 7) | ((k >> 1) & 8), kb = (k >> 3) & 1;
    w1f[(((c * NT1 + (n >> 4)) * 32) + kb * 16 + (n & 15)) * 16 + j] =
        (_Float16)W1[idx];
  }
  // ---- stage W2 [256,129] ----
  for (int idx = tid; idx < SDF_DIM * OUT_CH; idx += 512) {
    const int k = idx / OUT_CH, n = idx - k * OUT_CH;
    const int c = k >> 5, j = (k & 7) | ((k >> 1) & 8), kb = (k >> 3) & 1;
    w2f[(((c * NT2 + (n >> 4)) * 32) + kb * 16 + (n & 15)) * 16 + j] =
        (_Float16)W2[idx];
  }

  // ---- features: 2 lanes per point, 16 points per wave ----
  const int p   = lane >> 1;
  const int sub = lane & 1;
  const int pt  = blockIdx.x * PTS_PER_BLOCK + wave * 16 + p;
  float xn[3];
#pragma unroll
  for (int a = 0; a < 3; ++a) xn[a] = 2.f * xyz[3 * pt + a] - 1.f;

  const int M0[3] = {0, 0, 1}, M1[3] = {1, 2, 2}, VM[3] = {2, 1, 0};
#pragma unroll
  for (int i = 0; i < 3; ++i) {
    const float px = (xn[M0[i]] + 1.f) * 0.5f * (float)(GRIDW - 1);
    const float py = (xn[M1[i]] + 1.f) * 0.5f * (float)(GRIDW - 1);
    const float fx = fminf(fmaxf(floorf(px), 0.f), (float)(GRIDW - 2));
    const float fy = fminf(fmaxf(floorf(py), 0.f), (float)(GRIDW - 2));
    const float wx = px - fx, wy = py - fy;
    const int o00 = (int)fy * GRIDW + (int)fx;
    const float pl = (xn[VM[i]] + 1.f) * 0.5f * (float)(GRIDW - 1);
    const float fl = fminf(fmaxf(floorf(pl), 0.f), (float)(GRIDW - 2));
    const float wl = pl - fl;
    const int l0 = (int)fl;
    const int c0 = sub * 18;                      // lane handles 18 channels
    if (TP) {
      const _Float16* tpb = tpp + (size_t)(i * PLANE_PIX + o00) * NCOMP;
      const _Float16* tlb = tpl + (size_t)(i * GRIDW + l0) * NCOMP;
      for (int c = c0; c < c0 + 18; ++c) {
        const float f00 = (float)tpb[c];
        const float f01 = (float)tpb[NCOMP + c];
        const float f10 = (float)tpb[GRIDW * NCOMP + c];
        const float f11 = (float)tpb[(GRIDW + 1) * NCOMP + c];
        const float top = f00 * (1.f - wx) + f01 * wx;
        const float bot = f10 * (1.f - wx) + f11 * wx;
        const float bil = top * (1.f - wy) + bot * wy;
        const float lin = (float)tlb[c] * (1.f - wl) + (float)tlb[NCOMP + c] * wl;
        stg[p * STG_STRIDE + i * NCOMP + c] = (_Float16)(bil * lin);
      }
    } else {
      const float* pb = planes + (size_t)(i * NCOMP) * PLANE_PIX + o00;
      const float* lb = lines + (i * NCOMP) * GRIDW + l0;
      for (int c = c0; c < c0 + 18; ++c) {
        const float* pc = pb + (size_t)c * PLANE_PIX;
        const float f00 = pc[0], f01 = pc[1], f10 = pc[GRIDW], f11 = pc[GRIDW + 1];
        const float top = f00 * (1.f - wx) + f01 * wx;
        const float bot = f10 * (1.f - wx) + f11 * wx;
        const float bil = top * (1.f - wy) + bot * wy;
        const float* lc = lb + (size_t)c * GRIDW;
        const float lin = lc[0] * (1.f - wl) + lc[1] * wl;
        stg[p * STG_STRIDE + i * NCOMP + c] = (_Float16)(bil * lin);
      }
    }
  }
  // positional encoding (cols 108..128) and K-padding zeros (129..159)
  if (sub == 0) {
    _Float16* r = stg + p * STG_STRIDE + 108;
#pragma unroll
    for (int a = 0; a < 3; ++a) r[a] = (_Float16)xn[a];
    int o = 3;
#pragma unroll
    for (int f = 0; f < 3; ++f) {
      const float fr = (float)(1 << f);
#pragma unroll
      for (int a = 0; a < 3; ++a) r[o + a] = (_Float16)__sinf(xn[a] * fr);
#pragma unroll
      for (int a = 0; a < 3; ++a) r[o + 3 + a] = (_Float16)__cosf(xn[a] * fr);
      o += 6;
    }
  } else {
    _Float16* r = stg + p * STG_STRIDE;
    for (int c = IN_CH; c < KPAD1; ++c) r[c] = (_Float16)0.f;
  }
  __syncthreads();

  // ---- GEMM1: [16,160] @ [160,256]. 4 column tiles in flight (hazard-free
  //      WMMA stream) + double-buffered B fragments so the LDS loads for
  //      K-chunk c+1 are issued before the WMMAs of chunk c (partial
  //      s_wait_dscnt instead of full serialization). ----
  const int m   = lane & 15;                 // A row / C column-in-tile
  const int kb0 = (lane < 16) ? 0 : 8;       // A K-run base; D row half
  v16h aF[KC1];
#pragma unroll
  for (int c = 0; c < KC1; ++c) {
    const _Float16* ap = stg + m * STG_STRIDE + c * 32 + kb0;
    aF[c] = combine16(*(const v8h*)ap, *(const v8h*)(ap + 16));
  }
  for (int t0 = 0; t0 < NT1; t0 += 4) {
    v8f acc[4];
#pragma unroll
    for (int u = 0; u < 4; ++u) {
      const float bv = b1[(t0 + u) * 16 + m];
#pragma unroll
      for (int r = 0; r < 8; ++r) acc[u][r] = bv;
    }
    v16h bq[2][4];
#pragma unroll
    for (int u = 0; u < 4; ++u) {
      const _Float16* bp = w1f + ((0 * NT1 + t0 + u) * 32 + lane) * 16;
      bq[0][u] = combine16(*(const v8h*)bp, *(const v8h*)(bp + 8));
    }
#pragma unroll
    for (int c = 0; c < KC1; ++c) {
      if (c + 1 < KC1) {
#pragma unroll
        for (int u = 0; u < 4; ++u) {
          const _Float16* bp = w1f + (((c + 1) * NT1 + t0 + u) * 32 + lane) * 16;
          bq[(c + 1) & 1][u] = combine16(*(const v8h*)bp, *(const v8h*)(bp + 8));
        }
      }
#pragma unroll
      for (int u = 0; u < 4; ++u)
        acc[u] = __builtin_amdgcn_wmma_f32_16x16x32_f16(
            false, aF[c], false, bq[c & 1][u], (short)0, acc[u], false, false);
    }
    // softplus(100x)/100, restripe f32 D -> f16 A layout through LDS
#pragma unroll
    for (int u = 0; u < 4; ++u) {
#pragma unroll
      for (int r = 0; r < 8; ++r) {
        const float tv = 100.f * acc[u][r];
        const float sp = fmaxf(tv, 0.f) + log1pf(__expf(-fabsf(tv)));
        stg[(kb0 + r) * STG_STRIDE + (t0 + u) * 16 + m] = (_Float16)(sp * 0.01f);
      }
    }
  }
  __syncthreads();

  // ---- GEMM2: [16,256] @ [256,144] -> out[16,129], 3 tiles in flight,
  //      double-buffered B fragments ----
  v16h aG[KC2];
#pragma unroll
  for (int c = 0; c < KC2; ++c) {
    const _Float16* ap = stg + m * STG_STRIDE + c * 32 + kb0;
    aG[c] = combine16(*(const v8h*)ap, *(const v8h*)(ap + 16));
  }
  const int ptRow = blockIdx.x * PTS_PER_BLOCK + wave * 16;
  for (int t0 = 0; t0 < NT2; t0 += 3) {
    v8f acc[3];
#pragma unroll
    for (int u = 0; u < 3; ++u) {
      const int n = (t0 + u) * 16 + m;
      const float bv = (n < OUT_CH) ? b2[n] : 0.f;
#pragma unroll
      for (int r = 0; r < 8; ++r) acc[u][r] = bv;
    }
    v16h bq[2][3];
#pragma unroll
    for (int u = 0; u < 3; ++u) {
      const _Float16* bp = w2f + ((0 * NT2 + t0 + u) * 32 + lane) * 16;
      bq[0][u] = combine16(*(const v8h*)bp, *(const v8h*)(bp + 8));
    }
#pragma unroll
    for (int c = 0; c < KC2; ++c) {
      if (c + 1 < KC2) {
#pragma unroll
        for (int u = 0; u < 3; ++u) {
          const _Float16* bp = w2f + (((c + 1) * NT2 + t0 + u) * 32 + lane) * 16;
          bq[(c + 1) & 1][u] = combine16(*(const v8h*)bp, *(const v8h*)(bp + 8));
        }
      }
#pragma unroll
      for (int u = 0; u < 3; ++u)
        acc[u] = __builtin_amdgcn_wmma_f32_16x16x32_f16(
            false, aG[c], false, bq[c & 1][u], (short)0, acc[u], false, false);
    }
#pragma unroll
    for (int u = 0; u < 3; ++u) {
      const int n = (t0 + u) * 16 + m;
      if (n < OUT_CH) {
#pragma unroll
        for (int r = 0; r < 8; ++r)
          out[(size_t)(ptRow + kb0 + r) * OUT_CH + n] = acc[u][r];
      }
    }
  }
}

extern "C" void kernel_launch(void* const* d_in, const int* in_sizes, int n_in,
                              void* d_out, int out_size, void* d_ws, size_t ws_size,
                              hipStream_t stream) {
  const float* xyz    = (const float*)d_in[0];
  const float* planes = (const float*)d_in[1];
  const float* lines  = (const float*)d_in[2];
  const float* W1     = (const float*)d_in[3];
  const float* b1     = (const float*)d_in[4];
  const float* W2     = (const float*)d_in[5];
  const float* b2     = (const float*)d_in[6];
  float* out = (float*)d_out;

  const size_t tpPlaneHalves = (size_t)3 * PLANE_PIX * NCOMP;
  const size_t tpLineHalves  = (size_t)3 * GRIDW * NCOMP;
  const size_t tpBytes = (tpPlaneHalves + tpLineHalves) * sizeof(_Float16);
  const int blocks = NPTS / PTS_PER_BLOCK;

  if (ws_size >= tpBytes) {
    _Float16* tpp = (_Float16*)d_ws;
    _Float16* tpl = tpp + tpPlaneHalves;
    tp_planes_kernel<<<3 * (PLANE_PIX / 64), 256, 0, stream>>>(planes, tpp);
    tp_lines_kernel<<<(3 * NCOMP * GRIDW + 255) / 256, 256, 0, stream>>>(lines, tpl);
    hipFuncSetAttribute((const void*)tenso_fused_kernel<true>,
                        hipFuncAttributeMaxDynamicSharedMemorySize, SMEM_BYTES);
    tenso_fused_kernel<true><<<blocks, 512, SMEM_BYTES, stream>>>(
        xyz, planes, lines, tpp, tpl, W1, b1, W2, b2, out);
  } else {
    hipFuncSetAttribute((const void*)tenso_fused_kernel<false>,
                        hipFuncAttributeMaxDynamicSharedMemorySize, SMEM_BYTES);
    tenso_fused_kernel<false><<<blocks, 512, SMEM_BYTES, stream>>>(
        xyz, planes, lines, nullptr, nullptr, W1, b1, W2, b2, out);
  }
}
